// EucNormLoss_48344151884111
// MI455X (gfx1250) — matrix operand order repaired and checked by reading, not compiled, not run on hardware
//
#include <hip/hip_runtime.h>
#include <hip/hip_bf16.h>

#define NROWS 16384
#define DFEAT 128
#define NCLS  100

typedef _Float16 v4h  __attribute__((ext_vector_type(4)));
typedef _Float16 v8h  __attribute__((ext_vector_type(8)));
typedef _Float16 v16h __attribute__((ext_vector_type(16)));
typedef float    v8f  __attribute__((ext_vector_type(8)));

// raw v_sqrt_f32 (args are in [0,4]; no IEEE fixup sequence needed)
__device__ inline float fast_sqrtf(float x) {
#if __has_builtin(__builtin_amdgcn_sqrtf)
    return __builtin_amdgcn_sqrtf(x);
#else
    return sqrtf(x);
#endif
}

// low 32 bits of a generic pointer to LDS == LDS byte offset (aperture rule)
__device__ inline unsigned lds_addr32(const void* p) {
    return (unsigned)(size_t)p;
}

// ---------------- init: zero rowsum / class histogram / output ----------------
__global__ void EucNorm_init_kernel(float* rowsum, int* classcnt, float* out) {
    int i = blockIdx.x * blockDim.x + threadIdx.x;
    if (i < NROWS) rowsum[i] = 0.0f;
    if (i < 128)   classcnt[i] = 0;
    if (i == 0)    out[0] = 0.0f;
}

// ---------------- class histogram (denominator of per-row mean) ----------------
__global__ void EucNorm_hist_kernel(const int* __restrict__ labels, int* classcnt) {
    int i = blockIdx.x * blockDim.x + threadIdx.x;
    if (i < NROWS) atomicAdd(&classcnt[labels[i]], 1);
}

// ---------------- row L2-normalize, f32 -> f16 (one wave32 per row) ----------------
__global__ __launch_bounds__(256) void EucNorm_normalize_kernel(
        const float* __restrict__ feat, _Float16* __restrict__ feat16) {
    int row  = (blockIdx.x * blockDim.x + threadIdx.x) >> 5;   // wave id == row
    int lane = threadIdx.x & 31;
    if (row >= NROWS) return;
    const float4 v = *((const float4*)(feat + (size_t)row * DFEAT) + lane);
    float ss = v.x * v.x + v.y * v.y + v.z * v.z + v.w * v.w;
    ss += __shfl_xor(ss, 16, 32);
    ss += __shfl_xor(ss, 8, 32);
    ss += __shfl_xor(ss, 4, 32);
    ss += __shfl_xor(ss, 2, 32);
    ss += __shfl_xor(ss, 1, 32);
    float s = 1.0f / fmaxf(sqrtf(ss), 1e-12f);
    v4h h;
    h[0] = (_Float16)(v.x * s);
    h[1] = (_Float16)(v.y * s);
    h[2] = (_Float16)(v.z * s);
    h[3] = (_Float16)(v.w * s);
    *((v4h*)(feat16 + (size_t)row * DFEAT) + lane) = h;
}

// ---------------- main: WMMA Gram tile -> masked distance -> per-row sums ----------------
// Block = 256 threads = 8 wave32; block tile = 256 (i) x 128 (j).
// Each wave owns TWO 16-row i-strips: the same B fragments feed 8 WMMAs
// (two independent accumulator chains), halving B-fragment LDS traffic per WMMA.
// j-tile staged ROW-MAJOR in LDS via async DMA; B fragments via ds_load_b128,
// double-buffered and pipelined with DScnt-threshold waits (DS returns in order).
// Accumulators are fenced every 2nd iteration to bound live C tiles (stay <256 VGPRs).
#define BKSTRIDE 136   // halfs per LDS row: 272 B, 16B-aligned, staggers banks
__global__ __launch_bounds__(256) void EucNorm_gram_kernel(
        const _Float16* __restrict__ feat16,
        const int* __restrict__ labels,
        float* __restrict__ rowsum) {
    __shared__ _Float16 Bsh[128][BKSTRIDE]; // row-major [n][k] (~34 KB)
    __shared__ int jLab[128];
    __shared__ int iLab[256];

    const int tid    = threadIdx.x;
    const int jBase  = blockIdx.x * 128;
    const int iBase0 = blockIdx.y * 256;

    // stage labels
    if (tid < 128) {
        jLab[tid] = labels[jBase + tid];
    } else {
        const int t = tid - 128;
        iLab[2 * t]     = labels[iBase0 + 2 * t];
        iLab[2 * t + 1] = labels[iBase0 + 2 * t + 1];
    }

    // stage j-tile row-major with async global->LDS b128 copies:
    // 2 threads per feature row, 64 halfs (8 x 16B) each
    {
        const int row = tid >> 1;
        const int k0  = (tid & 1) * 64;
        const _Float16* src = feat16 + (size_t)(jBase + row) * DFEAT + k0;
        const unsigned dst = lds_addr32(&Bsh[row][k0]);   // 32-bit LDS byte addr
        #pragma unroll
        for (int v = 0; v < 8; ++v) {
            asm volatile("global_load_async_to_lds_b128 %0, %1, off offset:%2"
                         :
                         : "v"(dst), "v"(src), "i"(v * 16)
                         : "memory");
        }
        asm volatile("s_wait_asynccnt 0x0" ::: "memory");
    }
    __syncthreads();

    const int wave = tid >> 5;
    const int lane = tid & 31;
    const int hi   = lane >> 4;      // 0: lanes 0-15, 1: lanes 16-31
    const int l15  = lane & 15;
    const int iBaseA = iBase0 + wave * 32;   // this wave's 32-row strip pair

    // A fragments (16x32 f16 each), ISA layout:
    //   lanes 0-15:  M=l15, K = kk*32 + {0..7, 16..23}
    //   lanes 16-31: M=l15, K = kk*32 + {8..15, 24..31}
    v16h a0[4], a1[4];
    {
        const _Float16* ap0 = feat16 + (size_t)(iBaseA + l15) * DFEAT;
        const _Float16* ap1 = feat16 + (size_t)(iBaseA + 16 + l15) * DFEAT;
        #pragma unroll
        for (int kk = 0; kk < 4; ++kk) {
            const int k0 = kk * 32 + (hi ? 8 : 0);
            v8h lo0 = *(const v8h*)(ap0 + k0);
            v8h hh0 = *(const v8h*)(ap0 + k0 + 16);
            v8h lo1 = *(const v8h*)(ap1 + k0);
            v8h hh1 = *(const v8h*)(ap1 + k0 + 16);
            #pragma unroll
            for (int e = 0; e < 8; ++e) {
                a0[kk][e] = lo0[e]; a0[kk][8 + e] = hh0[e];
                a1[kk][e] = lo1[e]; a1[kk][8 + e] = hh1[e];
            }
        }
    }

    // hoist labels into registers (no LDS traffic inside the jj loop)
    int li0[8], li1[8], lj[8];
    #pragma unroll
    for (int r = 0; r < 8; ++r) {
        li0[r] = iLab[wave * 32 + 8 * hi + r];
        li1[r] = iLab[wave * 32 + 16 + 8 * hi + r];
    }
    #pragma unroll
    for (int t = 0; t < 8; ++t) lj[t] = jLab[t * 16 + l15];

    // pin label regs: their LDS reads (and the compiler's DS waits) retire
    // before the untracked pipelined ds_load_b128 sequence begins
    asm volatile(""
        : "+v"(li0[0]), "+v"(li0[1]), "+v"(li0[2]), "+v"(li0[3]),
          "+v"(li0[4]), "+v"(li0[5]), "+v"(li0[6]), "+v"(li0[7]),
          "+v"(li1[0]), "+v"(li1[1]), "+v"(li1[2]), "+v"(li1[3]),
          "+v"(li1[4]), "+v"(li1[5]), "+v"(li1[6]), "+v"(li1[7]),
          "+v"(lj[0]), "+v"(lj[1]), "+v"(lj[2]), "+v"(lj[3]),
          "+v"(lj[4]), "+v"(lj[5]), "+v"(lj[6]), "+v"(lj[7]));

    float acc0[8], acc1[8];
    #pragma unroll
    for (int r = 0; r < 8; ++r) { acc0[r] = 0.0f; acc1[r] = 0.0f; }

    // per-lane 16B half-row base inside the j-tile (row l15, col-half hi)
    const unsigned bBase    = lds_addr32(&Bsh[l15][hi * 8]);
    const unsigned jjStride = (unsigned)(16 * BKSTRIDE * 2);   // bytes per 16 j-rows

    v8h buf[2][8];   // double-buffered B fragments (8 x 16x16 tiles each)

    // prologue: issue jj=0 fragment loads
    asm volatile(
        "ds_load_b128 %0, %8 offset:0\n\t"
        "ds_load_b128 %1, %8 offset:32\n\t"
        "ds_load_b128 %2, %8 offset:64\n\t"
        "ds_load_b128 %3, %8 offset:96\n\t"
        "ds_load_b128 %4, %8 offset:128\n\t"
        "ds_load_b128 %5, %8 offset:160\n\t"
        "ds_load_b128 %6, %8 offset:192\n\t"
        "ds_load_b128 %7, %8 offset:224"
        : "=v"(buf[0][0]), "=v"(buf[0][1]), "=v"(buf[0][2]), "=v"(buf[0][3]),
          "=v"(buf[0][4]), "=v"(buf[0][5]), "=v"(buf[0][6]), "=v"(buf[0][7])
        : "v"(bBase));

    #pragma unroll
    for (int jj = 0; jj < 8; ++jj) {
        v8h* cb = buf[jj & 1];
        v8h* nb = buf[(jj + 1) & 1];

        if (jj < 7) {
            // issue next-iteration loads, then wait for current buffer only:
            // DS is in-order, so DScnt <= 8 means the 8 older loads are done.
            const unsigned na = bBase + (unsigned)(jj + 1) * jjStride;
            asm volatile(
                "ds_load_b128 %0, %16 offset:0\n\t"
                "ds_load_b128 %1, %16 offset:32\n\t"
                "ds_load_b128 %2, %16 offset:64\n\t"
                "ds_load_b128 %3, %16 offset:96\n\t"
                "ds_load_b128 %4, %16 offset:128\n\t"
                "ds_load_b128 %5, %16 offset:160\n\t"
                "ds_load_b128 %6, %16 offset:192\n\t"
                "ds_load_b128 %7, %16 offset:224\n\t"
                "s_wait_dscnt 0x8"
                : "=v"(nb[0]), "=v"(nb[1]), "=v"(nb[2]), "=v"(nb[3]),
                  "=v"(nb[4]), "=v"(nb[5]), "=v"(nb[6]), "=v"(nb[7]),
                  "+v"(cb[0]), "+v"(cb[1]), "+v"(cb[2]), "+v"(cb[3]),
                  "+v"(cb[4]), "+v"(cb[5]), "+v"(cb[6]), "+v"(cb[7])
                : "v"(na));
        } else {
            asm volatile("s_wait_dscnt 0x0"
                : "+v"(cb[0]), "+v"(cb[1]), "+v"(cb[2]), "+v"(cb[3]),
                  "+v"(cb[4]), "+v"(cb[5]), "+v"(cb[6]), "+v"(cb[7]));
        }

        // 8 WMMAs on the current buffer: two independent accumulator chains
        // (strips i and i+16) reuse each B fragment twice
        v8f c0 = {}, c1 = {};
        #pragma unroll
        for (int kk = 0; kk < 4; ++kk) {
            v16h b;
            #pragma unroll
            for (int e = 0; e < 8; ++e) {
                b[e]     = cb[2 * kk][e];
                b[8 + e] = cb[2 * kk + 1][e];
            }
            c0 = __builtin_amdgcn_wmma_f32_16x16x32_f16(
                    false, a0[kk], false, b, (short)0, c0, false, false);
            c1 = __builtin_amdgcn_wmma_f32_16x16x32_f16(
                    false, a1[kk], false, b, (short)0, c1, false, false);
        }

        // C layout: VGPR r -> M = r + 8*hi, N = l15
        #pragma unroll
        for (int r = 0; r < 8; ++r) {
            const float sq0 = fmaxf(2.0f - 2.0f * c0[r], 0.0f);
            const float d0  = fast_sqrtf(sq0);      // sqrt(0)==0: no guard needed
            acc0[r] += (li0[r] == lj[jj]) ? d0 : 0.0f;
            const float sq1 = fmaxf(2.0f - 2.0f * c1[r], 0.0f);
            const float d1  = fast_sqrtf(sq1);
            acc1[r] += (li1[r] == lj[jj]) ? d1 : 0.0f;
        }

        // scheduling fence every 2nd iteration: force C tiles dead so register
        // pressure stays under the 256-VGPR window (no s_set_vgpr_msb churn),
        // while still letting even-jj epilogues overlap odd-jj WMMAs
        if (jj & 1) {
            asm volatile(""
                : "+v"(acc0[0]), "+v"(acc0[1]), "+v"(acc0[2]), "+v"(acc0[3]),
                  "+v"(acc0[4]), "+v"(acc0[5]), "+v"(acc0[6]), "+v"(acc0[7]),
                  "+v"(acc1[0]), "+v"(acc1[1]), "+v"(acc1[2]), "+v"(acc1[3]),
                  "+v"(acc1[4]), "+v"(acc1[5]), "+v"(acc1[6]), "+v"(acc1[7]));
        }
    }

    // reduce over the 16 N-lanes within each half-wave, then one atomic per row
    #pragma unroll
    for (int r = 0; r < 8; ++r) {
        float s0 = acc0[r];
        s0 += __shfl_xor(s0, 1, 32);
        s0 += __shfl_xor(s0, 2, 32);
        s0 += __shfl_xor(s0, 4, 32);
        s0 += __shfl_xor(s0, 8, 32);
        float s1 = acc1[r];
        s1 += __shfl_xor(s1, 1, 32);
        s1 += __shfl_xor(s1, 2, 32);
        s1 += __shfl_xor(s1, 4, 32);
        s1 += __shfl_xor(s1, 8, 32);
        if (l15 == 0) {
            atomicAdd(&rowsum[iBaseA + r + 8 * hi], s0);
            atomicAdd(&rowsum[iBaseA + 16 + r + 8 * hi], s1);
        }
    }
}

// ---------------- final scalar: mean_i( rowsum[i] / count(label[i]) ) ----------------
__global__ __launch_bounds__(256) void EucNorm_finalize_kernel(
        const float* __restrict__ rowsum, const int* __restrict__ labels,
        const int* __restrict__ classcnt, float* out) {
    __shared__ float red[256];
    float s = 0.0f;
    for (int i = threadIdx.x; i < NROWS; i += 256)
        s += rowsum[i] / (float)classcnt[labels[i]];
    red[threadIdx.x] = s;
    __syncthreads();
    for (int off = 128; off > 0; off >>= 1) {
        if (threadIdx.x < off) red[threadIdx.x] += red[threadIdx.x + off];
        __syncthreads();
    }
    if (threadIdx.x == 0) out[0] = red[0] / (float)NROWS;
}

extern "C" void kernel_launch(void* const* d_in, const int* in_sizes, int n_in,
                              void* d_out, int out_size, void* d_ws, size_t ws_size,
                              hipStream_t stream) {
    const float* feat   = (const float*)d_in[0];
    const int*   labels = (const int*)d_in[1];
    float*       out    = (float*)d_out;

    char* ws = (char*)d_ws;
    _Float16* feat16   = (_Float16*)ws;                                   // 4 MB
    float*    rowsum   = (float*)(ws + (size_t)NROWS * DFEAT * 2);        // 64 KB
    int*      classcnt = (int*)(ws + (size_t)NROWS * DFEAT * 2
                                   + (size_t)NROWS * sizeof(float));      // 512 B

    hipLaunchKernelGGL(EucNorm_init_kernel, dim3((NROWS + 255) / 256), dim3(256),
                       0, stream, rowsum, classcnt, out);
    hipLaunchKernelGGL(EucNorm_hist_kernel, dim3(NROWS / 256), dim3(256),
                       0, stream, labels, classcnt);
    hipLaunchKernelGGL(EucNorm_normalize_kernel, dim3(NROWS / 8), dim3(256),
                       0, stream, feat, feat16);
    hipLaunchKernelGGL(EucNorm_gram_kernel, dim3(128, 64), dim3(256),
                       0, stream, feat16, labels, rowsum);
    hipLaunchKernelGGL(EucNorm_finalize_kernel, dim3(1), dim3(256),
                       0, stream, rowsum, labels, classcnt, out);
}
